// RWKV_Tmix_x070_32512902431314
// MI455X (gfx1250) — compile-verified, matrix operand order checked by reading, not run on hardware
//
#include <hip/hip_runtime.h>

// ---------------------------------------------------------------------------
// RWKV-7 Tmix forward for MI455X (gfx1250, wave32, WMMA bf16,
// async global->LDS double-buffered GEMM staging)
// B=8 T=1024 C=2048 H=32 N=64
// ---------------------------------------------------------------------------

constexpr int B_ = 8;
constexpr int T_ = 1024;
constexpr int C_ = 2048;
constexpr int H_ = 32;
constexpr int N_ = 64;
constexpr int M_ = B_ * T_;            // 8192 rows
constexpr int D_DECAY = 96, D_AAA = 96, D_MV = 64, D_GATE = 128;
constexpr float LN_EPS_ = 1e-5f * 64.0f;

typedef __attribute__((ext_vector_type(16))) __bf16 v16bf;
typedef __attribute__((ext_vector_type(8)))  float  v8f;

__device__ __forceinline__ float bf2f(unsigned short h) {
    unsigned int u = ((unsigned int)h) << 16;
    float f; __builtin_memcpy(&f, &u, 4); return f;
}
__device__ __forceinline__ unsigned short f2bf(float f) {
    unsigned int u; __builtin_memcpy(&u, &f, 4);
    unsigned int r = u + 0x7FFFu + ((u >> 16) & 1u);   // round-to-nearest-even
    return (unsigned short)(r >> 16);
}
__device__ __forceinline__ float sigm(float x) { return 1.0f / (1.0f + __expf(-x)); }

// CDNA5 async global->LDS copy (tracked by ASYNCcnt, completes in order).
// vdst = 32-bit LDS byte address, vaddr = 64-bit global address, saddr = off.
__device__ __forceinline__ void async_copy_b128(unsigned lds_off, unsigned long long gaddr) {
    asm volatile("global_load_async_to_lds_b128 %0, %1, off"
                 :: "v"(lds_off), "v"(gaddr) : "memory");
}

// ---------------------------------------------------------------------------
// 1) premix: x*=mask; xx = shift(x)-x; six lerps -> bf16
// ---------------------------------------------------------------------------
__global__ __launch_bounds__(256) void premix_kernel(
    const float* __restrict__ x, const float* __restrict__ mask,
    const float* __restrict__ mr, const float* __restrict__ mw,
    const float* __restrict__ mk, const float* __restrict__ mv,
    const float* __restrict__ ma, const float* __restrict__ mg,
    unsigned short* __restrict__ xr, unsigned short* __restrict__ xw,
    unsigned short* __restrict__ xk, unsigned short* __restrict__ xv,
    unsigned short* __restrict__ xa, unsigned short* __restrict__ xg)
{
    size_t idx = (size_t)blockIdx.x * 256 + threadIdx.x;   // over M_*C_
    int c = (int)(idx & (C_ - 1));
    size_t bt = idx >> 11;                                  // /C_
    int t = (int)(bt & (T_ - 1));
    float xm = x[idx] * mask[bt];
    float xp = (t > 0) ? x[idx - C_] * mask[bt - 1] : 0.0f;
    float xx = xp - xm;
    xr[idx] = f2bf(xm + xx * mr[c]);
    xw[idx] = f2bf(xm + xx * mw[c]);
    xk[idx] = f2bf(xm + xx * mk[c]);
    xv[idx] = f2bf(xm + xx * mv[c]);
    xa[idx] = f2bf(xm + xx * ma[c]);
    xg[idx] = f2bf(xm + xx * mg[c]);
}

// ---------------------------------------------------------------------------
// 2) f32 -> bf16 weight conversion
// ---------------------------------------------------------------------------
__global__ __launch_bounds__(256) void f2bf_kernel(const float* __restrict__ in,
                                                   unsigned short* __restrict__ out,
                                                   size_t n)
{
    size_t i = (size_t)blockIdx.x * 256 + threadIdx.x;
    if (i < n) out[i] = f2bf(in[i]);
}

// ---------------------------------------------------------------------------
// 3) WMMA bf16 NT-GEMM:  Cout[M,N] = A[M,K] * Wt[N,K]^T   (f32 accum)
//    Block tile 128x128, 8 waves (2x4), each wave 64x32 = 8 x (16x16) tiles.
//    K staged in 64-wide double-buffered LDS tiles filled by async copies.
// ---------------------------------------------------------------------------
constexpr int TM = 128, TN = 128, TKT = 64, TKP = TKT + 16;  // pitch 160B (16B aligned)

__global__ __launch_bounds__(256) void wmma_gemm_nt_kernel(
    const unsigned short* __restrict__ A,   // [M,K] bf16 row-major
    const unsigned short* __restrict__ Wt,  // [N,K] bf16 row-major
    float* __restrict__ Cout,               // [M,N] f32 row-major
    int M, int N, int K)
{
    __shared__ unsigned short As[2][TM][TKP];   // 40 KB
    __shared__ unsigned short Bs[2][TN][TKP];   // 40 KB

    const int bm = blockIdx.y * TM;
    const int bn = blockIdx.x * TN;
    const int tid = threadIdx.x;
    const int lane = tid & 31;
    const int wid = tid >> 5;              // 0..7
    const int wm = (wid >> 2) * 64;        // 0,64
    const int wn = (wid & 3) * 32;         // 0,32,64,96

    v8f acc[4][2] = {};

    const int mrow = lane & 15;
    const int kb = (lane >> 4) * 8;        // K-half select per ISA bf16 A/B layout

    // stage one 128x64 bf16 tile of A and of Wt into LDS buffer `buf`
    // (8 async b128 copies per thread, tracked on ASYNCcnt)
    auto stage = [&](int buf, int k0) {
        #pragma unroll
        for (int l = 0; l < 4; ++l) {
            int idx = tid + l * 256;       // 0..1023 chunks of 8 bf16
            int row = idx >> 3;
            int col = (idx & 7) * 8;
            async_copy_b128((unsigned)(size_t)&As[buf][row][col],
                            (unsigned long long)(const void*)(A  + (size_t)(bm + row) * K + k0 + col));
            async_copy_b128((unsigned)(size_t)&Bs[buf][row][col],
                            (unsigned long long)(const void*)(Wt + (size_t)(bn + row) * K + k0 + col));
        }
    };

    const int nk = K / TKT;
    stage(0, 0);

    for (int it = 0; it < nk; ++it) {
        const int cur = it & 1;
        const bool has_next = (it + 1) < nk;
        if (has_next) stage(cur ^ 1, (it + 1) * TKT);

        // async loads complete in order: waiting for <=8 outstanding means the
        // current tile's 8 copies have landed (only the prefetched 8 remain).
        if (has_next) asm volatile("s_wait_asynccnt 0x8" ::: "memory");
        else          asm volatile("s_wait_asynccnt 0x0" ::: "memory");
        __syncthreads();

        #pragma unroll
        for (int kk = 0; kk < TKT; kk += 32) {
            union Frag { v16bf v; unsigned int u[8]; };
            Frag af[4], bf[2];
            #pragma unroll
            for (int mi = 0; mi < 4; ++mi) {
                const unsigned short* ap = &As[cur][wm + mi * 16 + mrow][kk];
                #pragma unroll
                for (int e = 0; e < 4; ++e) {
                    af[mi].u[e]     = *(const unsigned int*)(ap + kb + 2 * e);       // K=kb..kb+7
                    af[mi].u[e + 4] = *(const unsigned int*)(ap + kb + 16 + 2 * e);  // K=kb+16..kb+23
                }
            }
            #pragma unroll
            for (int ni = 0; ni < 2; ++ni) {
                const unsigned short* bp = &Bs[cur][wn + ni * 16 + mrow][kk];
                #pragma unroll
                for (int e = 0; e < 4; ++e) {
                    bf[ni].u[e]     = *(const unsigned int*)(bp + kb + 2 * e);
                    bf[ni].u[e + 4] = *(const unsigned int*)(bp + kb + 16 + 2 * e);
                }
            }
            #pragma unroll
            for (int mi = 0; mi < 4; ++mi)
                #pragma unroll
                for (int ni = 0; ni < 2; ++ni)
                    acc[mi][ni] = __builtin_amdgcn_wmma_f32_16x16x32_bf16(
                        false, af[mi].v, false, bf[ni].v, (short)0, acc[mi][ni], false, false);
        }
        __syncthreads();   // all reads of buffer `cur` done before it is refilled
    }

    // C/D layout: VGPR v -> row (v + 8*(lane>>4)), col = lane&15
    const int colo = lane & 15;
    const int rhalf = (lane >> 4) * 8;
    #pragma unroll
    for (int mi = 0; mi < 4; ++mi)
        #pragma unroll
        for (int ni = 0; ni < 2; ++ni) {
            size_t rbase = (size_t)(bm + wm + mi * 16 + rhalf) * N + (bn + wn + ni * 16 + colo);
            #pragma unroll
            for (int v = 0; v < 8; ++v)
                Cout[rbase + (size_t)v * N] = acc[mi][ni][v];
        }
}

// ---------------------------------------------------------------------------
// 4) fused small MLPs + kk-normalize + k/v adjust (one block per row)
// ---------------------------------------------------------------------------
__global__ __launch_bounds__(128) void small_mlp_fuse_kernel(
    const unsigned short* __restrict__ xw, const unsigned short* __restrict__ xa,
    const unsigned short* __restrict__ xv, const unsigned short* __restrict__ xg,
    const float* __restrict__ w0, const float* __restrict__ w1, const float* __restrict__ w2,
    const float* __restrict__ a0, const float* __restrict__ a1, const float* __restrict__ a2,
    const float* __restrict__ v0, const float* __restrict__ v1, const float* __restrict__ v2,
    const float* __restrict__ g1, const float* __restrict__ g2,
    const float* __restrict__ k_k, const float* __restrict__ k_a,
    const float* __restrict__ mask, const float* __restrict__ v_first,
    float* __restrict__ kbuf,   // in: raw k, out: adjusted k
    float* __restrict__ vbuf,   // in: raw v, out: v'
    float* __restrict__ wdec, float* __restrict__ aab,
    float* __restrict__ bbb, float* __restrict__ gout)
{
    const int row = blockIdx.x;
    const size_t off = (size_t)row * C_;
    const int tid = threadIdx.x;

    __shared__ unsigned short sxw[C_], sxa[C_], sxv[C_], sxg[C_];
    __shared__ float hw[D_DECAY], ha[D_AAA], hv[D_MV], hg[D_GATE];
    __shared__ float s_a[C_], s_kk[C_], s_vm[C_];
    __shared__ float hns[H_];

    for (int c = tid; c < C_; c += 128) {
        sxw[c] = xw[off + c]; sxa[c] = xa[off + c];
        sxv[c] = xv[off + c]; sxg[c] = xg[off + c];
    }
    __syncthreads();

    // stage 1: 384 length-2048 dots (weight reads coalesced across threads)
    for (int job = tid; job < D_DECAY + D_AAA + D_MV + D_GATE; job += 128) {
        float acc = 0.f;
        if (job < D_DECAY) {
            for (int k = 0; k < C_; ++k) acc += bf2f(sxw[k]) * w1[(size_t)k * D_DECAY + job];
            hw[job] = tanhf(acc);
        } else if (job < D_DECAY + D_AAA) {
            int j = job - D_DECAY;
            for (int k = 0; k < C_; ++k) acc += bf2f(sxa[k]) * a1[(size_t)k * D_AAA + j];
            ha[j] = acc;
        } else if (job < D_DECAY + D_AAA + D_MV) {
            int j = job - D_DECAY - D_AAA;
            for (int k = 0; k < C_; ++k) acc += bf2f(sxv[k]) * v1[(size_t)k * D_MV + j];
            hv[j] = acc;
        } else {
            int j = job - D_DECAY - D_AAA - D_MV;
            for (int k = 0; k < C_; ++k) acc += bf2f(sxg[k]) * g1[(size_t)k * D_GATE + j];
            hg[j] = sigm(acc);
        }
    }
    __syncthreads();

    // stage 2: per-channel second projections + nonlinearities
    for (int c = tid; c < C_; c += 128) {
        float dw = 0.f, da = 0.f, dv = 0.f, dg = 0.f;
        for (int j = 0; j < D_DECAY; ++j) dw += hw[j] * w2[(size_t)j * C_ + c];
        for (int j = 0; j < D_AAA;   ++j) da += ha[j] * a2[(size_t)j * C_ + c];
        for (int j = 0; j < D_MV;    ++j) dv += hv[j] * v2[(size_t)j * C_ + c];
        for (int j = 0; j < D_GATE;  ++j) dg += hg[j] * g2[(size_t)j * C_ + c];
        float wraw = w0[c] + dw;
        float wv = -log1pf(__expf(-wraw)) - 0.5f;   // -softplus(-wraw) - 0.5
        wdec[off + c] = __expf(-__expf(wv));
        s_a[c]  = sigm(a0[c] + da);
        s_vm[c] = sigm(v0[c] + dv);
        gout[off + c] = dg;
        s_kk[c] = kbuf[off + c] * k_k[c];
    }
    __syncthreads();

    if (tid < H_) {
        float ss = 0.f;
        for (int i = 0; i < N_; ++i) { float q = s_kk[tid * N_ + i]; ss += q * q; }
        hns[tid] = fmaxf(sqrtf(ss), 1e-12f);
    }
    __syncthreads();

    const float mrow = mask[row];
    for (int c = tid; c < C_; c += 128) {
        int h = c >> 6;
        float kkn = s_kk[c] / hns[h];
        float av = s_a[c];
        aab[off + c] = -kkn;
        bbb[off + c] = kkn * av;
        float kr = kbuf[off + c];
        kbuf[off + c] = kr * (1.f + (av - 1.f) * k_a[c]);
        float vr = vbuf[off + c];
        vbuf[off + c] = (vr + (v_first[off + c] - vr) * s_vm[c]) * mrow;
    }
}

// ---------------------------------------------------------------------------
// 5) sequential RWKV-7 scan: one block per (b,h); thread j owns state row j
// ---------------------------------------------------------------------------
__global__ __launch_bounds__(64) void scan_kernel(
    const float* __restrict__ r, const float* __restrict__ wdec,
    const float* __restrict__ k, const float* __restrict__ v,
    const float* __restrict__ aa, const float* __restrict__ bb,
    float* __restrict__ y)
{
    const int bh = blockIdx.x;
    const int b = bh / H_, h = bh % H_;
    const int j = threadIdx.x;
    __shared__ float lr[N_], lw[N_], lk[N_], lv[N_], la[N_], lb[N_];

    float S[N_];
    #pragma unroll
    for (int i = 0; i < N_; ++i) S[i] = 0.f;

    size_t base = (size_t)b * T_ * C_ + (size_t)h * N_;
    for (int t = 0; t < T_; ++t) {
        size_t o = base + (size_t)t * C_;
        lr[j] = r[o + j]; lw[j] = wdec[o + j]; lk[j] = k[o + j];
        lv[j] = v[o + j]; la[j] = aa[o + j];   lb[j] = bb[o + j];
        __syncthreads();
        float sa = 0.f;
        #pragma unroll
        for (int i = 0; i < N_; ++i) sa += S[i] * la[i];
        const float vj = lv[j];
        float yj = 0.f;
        #pragma unroll
        for (int i = 0; i < N_; ++i) {
            float s = S[i] * lw[i] + sa * lb[i] + vj * lk[i];
            S[i] = s;
            yj += s * lr[i];
        }
        y[o + j] = yj;
        __syncthreads();
    }
}

// ---------------------------------------------------------------------------
// 6) GroupNorm + rk*v bonus + gate -> bf16 (one block per row)
// ---------------------------------------------------------------------------
__global__ __launch_bounds__(256) void gn_epilogue_kernel(
    const float* __restrict__ y, const float* __restrict__ r,
    const float* __restrict__ k, const float* __restrict__ v,
    const float* __restrict__ g, const float* __restrict__ r_k,
    const float* __restrict__ ln_w, const float* __restrict__ ln_b,
    unsigned short* __restrict__ yg)
{
    const int row = blockIdx.x;
    const size_t off = (size_t)row * C_;
    const int tid = threadIdx.x;
    __shared__ float sy[C_];
    __shared__ float hmu[H_], hrs[H_], hrk[H_];

    for (int c = tid; c < C_; c += 256) sy[c] = y[off + c];
    __syncthreads();

    if (tid < H_) {
        float s = 0.f, s2 = 0.f, rk = 0.f;
        for (int i = 0; i < N_; ++i) {
            float q = sy[tid * N_ + i];
            s += q; s2 += q * q;
            rk += r[off + tid * N_ + i] * k[off + tid * N_ + i] * r_k[tid * N_ + i];
        }
        float mu = s * (1.0f / N_);
        float var = s2 * (1.0f / N_) - mu * mu;
        hmu[tid] = mu; hrs[tid] = rsqrtf(var + LN_EPS_); hrk[tid] = rk;
    }
    __syncthreads();

    for (int c = tid; c < C_; c += 256) {
        int h = c >> 6;
        float yn = (sy[c] - hmu[h]) * hrs[h] * ln_w[c] + ln_b[c];
        yn += hrk[h] * v[off + c];
        yg[off + c] = f2bf(yn * g[off + c]);
    }
}

// ---------------------------------------------------------------------------
// launch
// ---------------------------------------------------------------------------
extern "C" void kernel_launch(void* const* d_in, const int* in_sizes, int n_in,
                              void* d_out, int out_size, void* d_ws, size_t ws_size,
                              hipStream_t stream)
{
    const float* x        = (const float*)d_in[0];
    const float* amask    = (const float*)d_in[1];
    const float* v_first  = (const float*)d_in[2];
    const float* x_r = (const float*)d_in[3], *x_w = (const float*)d_in[4];
    const float* x_k = (const float*)d_in[5], *x_v = (const float*)d_in[6];
    const float* x_a = (const float*)d_in[7], *x_g = (const float*)d_in[8];
    const float* w0 = (const float*)d_in[9],  *w1 = (const float*)d_in[10], *w2 = (const float*)d_in[11];
    const float* a0 = (const float*)d_in[12], *a1 = (const float*)d_in[13], *a2 = (const float*)d_in[14];
    const float* v0 = (const float*)d_in[15], *v1 = (const float*)d_in[16], *v2 = (const float*)d_in[17];
    const float* g1 = (const float*)d_in[18], *g2 = (const float*)d_in[19];
    const float* k_k = (const float*)d_in[20], *k_a = (const float*)d_in[21];
    const float* r_k = (const float*)d_in[22];
    const float* W_r = (const float*)d_in[23], *W_k = (const float*)d_in[24];
    const float* W_v = (const float*)d_in[25], *W_o = (const float*)d_in[26];
    const float* ln_w = (const float*)d_in[27], *ln_b = (const float*)d_in[28];
    float* out = (float*)d_out;

    const size_t MC = (size_t)M_ * C_;
    const size_t CC = (size_t)C_ * C_;

    // workspace carve-up
    char* p = (char*)d_ws;
    auto take = [&](size_t bytes) { char* q = p; p += (bytes + 255) & ~(size_t)255; return q; };
    unsigned short* xr = (unsigned short*)take(MC * 2);
    unsigned short* xw = (unsigned short*)take(MC * 2);
    unsigned short* xk = (unsigned short*)take(MC * 2);
    unsigned short* xv = (unsigned short*)take(MC * 2);
    unsigned short* xa = (unsigned short*)take(MC * 2);
    unsigned short* xg = (unsigned short*)take(MC * 2);
    unsigned short* Wr_bf = (unsigned short*)take(CC * 2);
    unsigned short* Wk_bf = (unsigned short*)take(CC * 2);
    unsigned short* Wv_bf = (unsigned short*)take(CC * 2);
    unsigned short* Wo_bf = (unsigned short*)take(CC * 2);
    float* rbuf = (float*)take(MC * 4);
    float* kbuf = (float*)take(MC * 4);
    float* vbuf = (float*)take(MC * 4);
    float* wdec = (float*)take(MC * 4);
    float* aab  = (float*)take(MC * 4);
    float* bbb  = (float*)take(MC * 4);
    float* gbuf = (float*)take(MC * 4);
    float* ybuf = (float*)take(MC * 4);
    unsigned short* yg = (unsigned short*)take(MC * 2);

    // 1) premix
    premix_kernel<<<dim3((unsigned)(MC / 256)), dim3(256), 0, stream>>>(
        x, amask, x_r, x_w, x_k, x_v, x_a, x_g, xr, xw, xk, xv, xa, xg);

    // 2) weight conversion
    unsigned wg = (unsigned)((CC + 255) / 256);
    f2bf_kernel<<<wg, 256, 0, stream>>>(W_r, Wr_bf, CC);
    f2bf_kernel<<<wg, 256, 0, stream>>>(W_k, Wk_bf, CC);
    f2bf_kernel<<<wg, 256, 0, stream>>>(W_v, Wv_bf, CC);
    f2bf_kernel<<<wg, 256, 0, stream>>>(W_o, Wo_bf, CC);

    // 3) big WMMA GEMMs: r, k, v
    dim3 ggrid(C_ / TN, M_ / TM);
    wmma_gemm_nt_kernel<<<ggrid, 256, 0, stream>>>(xr, Wr_bf, rbuf, M_, C_, C_);
    wmma_gemm_nt_kernel<<<ggrid, 256, 0, stream>>>(xk, Wk_bf, kbuf, M_, C_, C_);
    wmma_gemm_nt_kernel<<<ggrid, 256, 0, stream>>>(xv, Wv_bf, vbuf, M_, C_, C_);

    // 4) fused small MLPs + kk / k / v adjust
    small_mlp_fuse_kernel<<<M_, 128, 0, stream>>>(
        xw, xa, xv, xg, w0, w1, w2, a0, a1, a2, v0, v1, v2, g1, g2,
        k_k, k_a, amask, v_first, kbuf, vbuf, wdec, aab, bbb, gbuf);

    // 5) sequential scan
    scan_kernel<<<B_ * H_, N_, 0, stream>>>(rbuf, wdec, kbuf, vbuf, aab, bbb, ybuf);

    // 6) GroupNorm + rk*v + gate
    gn_epilogue_kernel<<<M_, 256, 0, stream>>>(
        ybuf, rbuf, kbuf, vbuf, gbuf, r_k, ln_w, ln_b, yg);

    // 7) output projection (WMMA) -> first output
    wmma_gemm_nt_kernel<<<ggrid, 256, 0, stream>>>(yg, Wo_bf, out, M_, C_, C_);

    // 8) second output = v_first passthrough
    hipMemcpyAsync(out + MC, v_first, MC * sizeof(float),
                   hipMemcpyDeviceToDevice, stream);
}
